// ParallelSelfAttention_85873576116361
// MI455X (gfx1250) — matrix operand order repaired and checked by reading, not compiled, not run on hardware
//
#include <hip/hip_runtime.h>
#include <hip/hip_bf16.h>

typedef __attribute__((ext_vector_type(16))) __bf16 v16bf;
typedef __attribute__((ext_vector_type(8)))  __bf16 v8bf;
typedef __attribute__((ext_vector_type(8)))  float  v8f;
typedef __attribute__((ext_vector_type(4)))  unsigned int v4u;
typedef __attribute__((ext_vector_type(8)))  int v8i;
typedef __attribute__((ext_vector_type(4)))  int v4i;

constexpr int S  = 2048;
constexpr int H  = 1024;
constexpr int NT = 1024;
constexpr int NH = 16;
constexpr int HD = 64;

#if defined(__has_builtin)
#if __has_builtin(__builtin_amdgcn_tensor_load_to_lds) && \
    __has_builtin(__builtin_amdgcn_s_wait_tensorcnt)
#define USE_TDM 1
#endif
#endif
#ifndef USE_TDM
#define USE_TDM 0
#endif

// ---------- small helpers ----------

__device__ inline __bf16 f2bf(float f) {
  union { float f; unsigned u; } c; c.f = f;
  unsigned u = c.u;
  unsigned r = u + 0x7fffu + ((u >> 16) & 1u);   // round to nearest even
  unsigned short hbits = (unsigned short)(r >> 16);
  union { unsigned short s; __bf16 b; } o; o.s = hbits;
  return o.b;
}

__device__ inline v8f zero8() {
  v8f z;
#pragma unroll
  for (int i = 0; i < 8; ++i) z[i] = 0.0f;
  return z;
}

// B-tile (K x 16, 16-bit): lane holds column n=l&15; lanes 0-15 hold K 0..15,
// lanes 16-31 hold K 16..31 -> contiguous 16 bf16 per lane.
__device__ inline v16bf loadB(const __bf16* p) { return *(const v16bf*)p; }

// A-tile (16 x 32, 16-bit): lane holds row m=l&15; lanes 0-15: K 0..15,
// lanes 16-31: K 16..31.
__device__ inline v16bf loadA32(const __bf16* rowp, int lane) {
  const int kb = (lane & 16) ? 16 : 0;
  v8bf lo = *(const v8bf*)(rowp + kb);
  v8bf hi = *(const v8bf*)(rowp + kb + 8);
  return __builtin_shufflevector(lo, hi, 0, 1, 2, 3, 4, 5, 6, 7,
                                        8, 9, 10, 11, 12, 13, 14, 15);
}

__device__ inline v8f wmma_bf16(v16bf a, v16bf b, v8f c) {
  return __builtin_amdgcn_wmma_f32_16x16x32_bf16(
      /*neg_a=*/false, a, /*neg_b=*/false, b,
      /*c_mod=*/(short)0, c, /*reuse_a=*/false, /*reuse_b=*/false);
}

__device__ inline float gelu_exact(float x) {
  return 0.5f * x * (1.0f + erff(x * 0.70710678118654752f));
}

// ---------- conversion / transpose kernels ----------

__global__ void cvt_f32_bf16(const float* __restrict__ src,
                             __bf16* __restrict__ dst, int n) {
  int i = blockIdx.x * blockDim.x + threadIdx.x;
  int stride = gridDim.x * blockDim.x;
  for (; i < n; i += stride) dst[i] = f2bf(src[i]);
}

// src: [NT][H] fp32 (value tokens) -> dst: [H][NT] bf16 (dst[d][t] = src[t][d])
__global__ void transpose_to_bf16(const float* __restrict__ src,
                                  __bf16* __restrict__ dst) {
  __shared__ float tile[32][33];
  const int bx = blockIdx.x * 32;   // d (col of src)
  const int by = blockIdx.y * 32;   // t (row of src)
  const int tx = threadIdx.x, ty = threadIdx.y;   // block (32, 8)
#pragma unroll
  for (int r = ty; r < 32; r += 8)
    tile[r][tx] = src[(by + r) * H + bx + tx];
  __syncthreads();
#pragma unroll
  for (int r = ty; r < 32; r += 8)
    dst[(bx + r) * NT + by + tx] = f2bf(tile[tx][r]);
}

// vb: [S][H] bf16 -> vhT: [NH][HD][S] bf16 (per-head transposed V)
__global__ void transpose_head(const __bf16* __restrict__ vb,
                               __bf16* __restrict__ vhT) {
  __shared__ __align__(16) __bf16 tile[64][65];
  const int h  = blockIdx.y;
  const int s0 = blockIdx.x * 64;
  const int t  = threadIdx.x;                     // 256 threads
  const int row = t >> 2, c0 = (t & 3) * 16;
#pragma unroll
  for (int i = 0; i < 16; ++i)
    tile[row][c0 + i] = vb[(s0 + row) * H + h * HD + c0 + i];
  __syncthreads();
  const int d = t >> 2, so = (t & 3) * 16;
#pragma unroll
  for (int i = 0; i < 16; ++i)
    vhT[(h * HD + d) * S + s0 + so + i] = tile[so + i][d];
}

// ---------- fused pattention: out = (gelu_l2norm(X @ KEY^T)) @ VAL ----------
// X: [S][H] bf16, KEY: [NT][H] bf16, VALT: [H][NT] bf16.
// 32 rows per workgroup (2 M-tiles per wave), 8 waves tile N.
// x rows are staged into LDS by the Tensor Data Mover; the w buffer overlays
// the x tile (x is dead after phase A; separated by the norm barrier).

struct PattArgs {
  const __bf16* key[3];
  const __bf16* valT[3];
  __bf16*       outB[3];
};

template <bool OUTF32>
__global__ __launch_bounds__(256)
void pattention_kernel(const __bf16* __restrict__ X, PattArgs args,
                       float* __restrict__ outF) {
  __shared__ __align__(32) __bf16 xa[32][H];   // x tile, later reused as w
  __shared__ float norms[32];

  const int t     = threadIdx.x;
  const int lane  = t & 31;
  const int wave  = t >> 5;
  const int m     = lane & 15;
  const int rbase = (lane & 16) >> 1;          // 0 or 8 (C-layout row base)
  const int bsel  = lane & 16;                 // 0 or 16 (B-layout K base)
  const int base  = blockIdx.x * 32;           // row block
  const int mat   = blockIdx.y;

  const __bf16* KEY  = args.key[mat];
  const __bf16* VALT = args.valT[mat];
  __bf16*       outB = args.outB[mat];

  if (t < 32) norms[t] = 0.0f;

  // ---- stage x rows [base, base+32) x H into LDS ----
#if USE_TDM
  if (wave == 0) {
    // D# per CDNA5 ISA sec.8: 2D tile, 1-byte elements.
    const unsigned lds_base = (unsigned)(uintptr_t)(void*)&xa[0][0];
    const unsigned long long ga =
        (unsigned long long)(uintptr_t)X + (unsigned long long)base * (H * 2);
    const unsigned rowb = (unsigned)(H * 2);   // 2048 bytes per row
    v4u g0;
    g0[0] = 1u;                                 // count=1 (valid descriptor)
    g0[1] = lds_base;                           // lds_addr
    g0[2] = (unsigned)(ga & 0xffffffffull);     // global_addr[31:0]
    g0[3] = (unsigned)((ga >> 32) & 0x1ffffffull) | (2u << 30);  // addr hi|type=2
    v8i g1;
    g1[0] = 0;                                  // wg_mask=0, data_size=1B
    g1[1] = (int)((rowb & 0xffffu) << 16);      // tensor_dim0[15:0] @63:48
    g1[2] = (int)(((unsigned)S & 0xffffu) << 16);  // tensor_dim1[15:0] @95:80
    g1[3] = (int)((rowb & 0xffffu) << 16);      // tile_dim0 @127:112
    g1[4] = 32;                                 // tile_dim1=32, tile_dim2=0
    g1[5] = (int)rowb;                          // tensor_dim0_stride[31:0]
    g1[6] = 0;
    g1[7] = 0;
    v4i gz4; gz4[0] = 0; gz4[1] = 0; gz4[2] = 0; gz4[3] = 0;
    v8i gz8;
#pragma unroll
    for (int i = 0; i < 8; ++i) gz8[i] = 0;
    // 6-arg form (clang-23 / therock headers): groups 0..3 + extra + cpol
    __builtin_amdgcn_tensor_load_to_lds(g0, g1, gz4, gz4, gz8, 0);
    __builtin_amdgcn_s_wait_tensorcnt(0);
  }
#else
  {
    const v16bf* src = (const v16bf*)(X + (size_t)base * H);
    v16bf* dst = (v16bf*)&xa[0][0];
#pragma unroll
    for (int i = 0; i < 8; ++i) dst[t + i * 256] = src[t + i * 256];
  }
#endif
  __syncthreads();

  // ---- phase A: attn = X @ KEY^T  (2 M-tiles x 8 N-tiles per wave) ----
  v8f acc[2][8];
#pragma unroll
  for (int mt = 0; mt < 2; ++mt)
#pragma unroll
    for (int nt = 0; nt < 8; ++nt) acc[mt][nt] = zero8();

  for (int kk = 0; kk < H; kk += 32) {
    v16bf a0 = loadA32(&xa[m][kk], lane);
    v16bf a1 = loadA32(&xa[16 + m][kk], lane);
#pragma unroll
    for (int nt = 0; nt < 8; ++nt) {
      const int col = wave * 128 + nt * 16 + m;
      const __bf16* bp = KEY + (size_t)col * H + kk + bsel;
      v16bf b = loadB(bp);
      if (kk + 256 < H) __builtin_prefetch(bp + 256, 0, 1);
      acc[0][nt] = wmma_bf16(a0, b, acc[0][nt]);
      acc[1][nt] = wmma_bf16(a1, b, acc[1][nt]);
    }
  }

  // ---- gelu + per-row sum of squares ----
  float rowsq[2][8];
#pragma unroll
  for (int mt = 0; mt < 2; ++mt)
#pragma unroll
    for (int e = 0; e < 8; ++e) rowsq[mt][e] = 0.0f;
#pragma unroll
  for (int mt = 0; mt < 2; ++mt) {
#pragma unroll
    for (int nt = 0; nt < 8; ++nt) {
#pragma unroll
      for (int e = 0; e < 8; ++e) {
        float g = gelu_exact(acc[mt][nt][e]);
        acc[mt][nt][e] = g;
        rowsq[mt][e] += g * g;
      }
    }
  }
#pragma unroll
  for (int mt = 0; mt < 2; ++mt) {
#pragma unroll
    for (int e = 0; e < 8; ++e) {
      float v = rowsq[mt][e];
      v += __shfl_xor(v, 1, 32);
      v += __shfl_xor(v, 2, 32);
      v += __shfl_xor(v, 4, 32);
      v += __shfl_xor(v, 8, 32);
      rowsq[mt][e] = v;
    }
  }
  if (m == 0) {                                   // lanes 0 and 16
#pragma unroll
    for (int mt = 0; mt < 2; ++mt)
#pragma unroll
      for (int e = 0; e < 8; ++e)
        atomicAdd(&norms[mt * 16 + e + rbase], rowsq[mt][e]);
  }
  __syncthreads();   // all waves past phase A: xa (x) is dead, norms complete

  // ---- scale w = g / ||g|| * sqrt(NT); overlay into xa as bf16 ----
#pragma unroll
  for (int mt = 0; mt < 2; ++mt) {
#pragma unroll
    for (int e = 0; e < 8; ++e) {
      const int r = mt * 16 + e + rbase;
      const float sc = 32.0f * rsqrtf(norms[r]);  // sqrt(1024) = 32
#pragma unroll
      for (int nt = 0; nt < 8; ++nt) {
        const int col = wave * 128 + nt * 16 + m;
        xa[r][col] = f2bf(acc[mt][nt][e] * sc);
      }
    }
  }
  __syncthreads();

  // ---- phase C: out = w @ VAL  (B from VALT, contiguous along K=t) ----
  v8f acc2[2][8];
#pragma unroll
  for (int mt = 0; mt < 2; ++mt)
#pragma unroll
    for (int nt = 0; nt < 8; ++nt) acc2[mt][nt] = zero8();

  for (int kk = 0; kk < NT; kk += 32) {
    v16bf a0 = loadA32(&xa[m][kk], lane);
    v16bf a1 = loadA32(&xa[16 + m][kk], lane);
#pragma unroll
    for (int nt = 0; nt < 8; ++nt) {
      const int col = wave * 128 + nt * 16 + m;
      const __bf16* bp = VALT + (size_t)col * NT + kk + bsel;
      v16bf b = loadB(bp);
      if (kk + 256 < NT) __builtin_prefetch(bp + 256, 0, 1);
      acc2[0][nt] = wmma_bf16(a0, b, acc2[0][nt]);
      acc2[1][nt] = wmma_bf16(a1, b, acc2[1][nt]);
    }
  }

#pragma unroll
  for (int mt = 0; mt < 2; ++mt) {
#pragma unroll
    for (int nt = 0; nt < 8; ++nt) {
#pragma unroll
      for (int e = 0; e < 8; ++e) {
        const int row = base + mt * 16 + e + rbase;
        const int col = wave * 128 + nt * 16 + m;
        if (OUTF32) outF[(size_t)row * H + col] = acc2[mt][nt][e];
        else        outB[(size_t)row * H + col] = f2bf(acc2[mt][nt][e]);
      }
    }
  }
}

// ---------- causal flash attention, one wave per 16-row query tile ----------
// Q,K: [S][H] bf16; VT: [NH][HD][S] bf16; CTX: [S][H] bf16.

__global__ __launch_bounds__(128)
void flash_attn(const __bf16* __restrict__ Q, const __bf16* __restrict__ K,
                const __bf16* __restrict__ VT, __bf16* __restrict__ CTX) {
  __shared__ __align__(16) __bf16 pbuf[4][16][32];   // per-wave P transpose bounce

  const int t     = threadIdx.x;
  const int lane  = t & 31;
  const int wave  = t >> 5;
  const int m     = lane & 15;
  const int rbase = (lane & 16) >> 1;
  const int bsel  = lane & 16;
  const int h     = blockIdx.y;
  const int qs    = blockIdx.x * 64 + wave * 16;
  const float scale = 0.125f;                        // 1/sqrt(64)

  const __bf16* qrowp = Q + (qs + m) * H + h * HD;
  v16bf aq0 = loadA32(qrowp, lane);
  v16bf aq1 = loadA32(qrowp + 32, lane);

  float mrow[8], lrow[8];
  v8f O[4];
#pragma unroll
  for (int e = 0; e < 8; ++e) { mrow[e] = -1e30f; lrow[e] = 0.0f; }
#pragma unroll
  for (int dt = 0; dt < 4; ++dt) O[dt] = zero8();

  const int kend = qs + 16;                          // causal key bound
  for (int k0 = 0; k0 < kend; k0 += 32) {
    v8f s0 = zero8(), s1 = zero8();
    const int kn0 = k0 + m;
    const int kn1 = kn0 + 16;
    const __bf16* kp0 = K + kn0 * H + h * HD + bsel;
    const __bf16* kp1 = K + kn1 * H + h * HD + bsel;
    v16bf b00 = loadB(kp0);        // hd 0..31 slice
    v16bf b01 = loadB(kp0 + 32);   // hd 32..63 slice
    v16bf b10 = loadB(kp1);
    v16bf b11 = loadB(kp1 + 32);
    s0 = wmma_bf16(aq0, b00, s0);  s0 = wmma_bf16(aq1, b01, s0);
    s1 = wmma_bf16(aq0, b10, s1);  s1 = wmma_bf16(aq1, b11, s1);

    // ---- online softmax update ----
#pragma unroll
    for (int e = 0; e < 8; ++e) {
      const int qrow = qs + e + rbase;
      float v0 = s0[e] * scale; if (k0 + m > qrow)      v0 = -1e30f;
      float v1 = s1[e] * scale; if (k0 + 16 + m > qrow) v1 = -1e30f;
      float rmx = fmaxf(v0, v1);
      rmx = fmaxf(rmx, __shfl_xor(rmx, 1, 32));
      rmx = fmaxf(rmx, __shfl_xor(rmx, 2, 32));
      rmx = fmaxf(rmx, __shfl_xor(rmx, 4, 32));
      rmx = fmaxf(rmx, __shfl_xor(rmx, 8, 32));
      const float mnew  = fmaxf(mrow[e], rmx);
      const float alpha = __expf(mrow[e] - mnew);
      const float p0 = __expf(v0 - mnew);
      const float p1 = __expf(v1 - mnew);
      float rs = p0 + p1;
      rs += __shfl_xor(rs, 1, 32);
      rs += __shfl_xor(rs, 2, 32);
      rs += __shfl_xor(rs, 4, 32);
      rs += __shfl_xor(rs, 8, 32);
      lrow[e] = lrow[e] * alpha + rs;
      mrow[e] = mnew;
#pragma unroll
      for (int dt = 0; dt < 4; ++dt) O[dt][e] *= alpha;
      pbuf[wave][e + rbase][m]      = f2bf(p0);
      pbuf[wave][e + rbase][m + 16] = f2bf(p1);
    }

    // wave-local LDS sync (waves diverge -> no __syncthreads here)
    asm volatile("s_wait_dscnt 0x0" ::: "memory");

    // ---- P (A-layout) @ V chunk ----
    v16bf ap = loadA32(&pbuf[wave][m][0], lane);
#pragma unroll
    for (int dt = 0; dt < 4; ++dt) {
      const int d = dt * 16 + m;
      v16bf bv = loadB(VT + (h * HD + d) * S + k0 + bsel);
      O[dt] = wmma_bf16(ap, bv, O[dt]);
    }
    asm volatile("s_wait_dscnt 0x0" ::: "memory");   // WAR guard vs next-iter writes
  }

  // ---- epilogue: normalize and store ctx (bf16) ----
#pragma unroll
  for (int e = 0; e < 8; ++e) {
    const float inv = 1.0f / lrow[e];
    const int row = qs + e + rbase;
#pragma unroll
    for (int dt = 0; dt < 4; ++dt)
      CTX[row * H + h * HD + dt * 16 + m] = f2bf(O[dt][e] * inv);
  }
}

// ---------- host launcher ----------

extern "C" void kernel_launch(void* const* d_in, const int* in_sizes, int n_in,
                              void* d_out, int out_size, void* d_ws, size_t ws_size,
                              hipStream_t stream) {
  (void)in_sizes; (void)n_in; (void)out_size;

  char* ws = (char*)d_ws;
  size_t off = 0;
  auto alloc = [&](size_t bytes) -> void* {
    void* p = ws + off;
    off += (bytes + 255) & ~(size_t)255;
    return p;
  };

  __bf16* xb = (__bf16*)alloc((size_t)S * H * 2);
  __bf16* keyb[4];
  __bf16* valT[4];
  for (int i = 0; i < 4; ++i) keyb[i] = (__bf16*)alloc((size_t)NT * H * 2);
  for (int i = 0; i < 4; ++i) valT[i] = (__bf16*)alloc((size_t)H * NT * 2);
  __bf16* qb  = (__bf16*)alloc((size_t)S * H * 2);
  __bf16* kb  = (__bf16*)alloc((size_t)S * H * 2);
  __bf16* vb  = (__bf16*)alloc((size_t)S * H * 2);
  __bf16* vhT = (__bf16*)alloc((size_t)NH * HD * S * 2);
  __bf16* ctx = (__bf16*)alloc((size_t)S * H * 2);
  if (off > ws_size) return;   // workspace too small; bail deterministically

  const float* x = (const float*)d_in[0];
  const int keyIdx[4] = {1, 3, 5, 7};   // q_key, k_key, v_key, proj_key
  const int valIdx[4] = {2, 4, 6, 8};   // q_val, k_val, v_val, proj_val

  cvt_f32_bf16<<<1024, 256, 0, stream>>>(x, xb, S * H);
  for (int i = 0; i < 4; ++i)
    cvt_f32_bf16<<<1024, 256, 0, stream>>>((const float*)d_in[keyIdx[i]],
                                           keyb[i], NT * H);
  for (int i = 0; i < 4; ++i)
    transpose_to_bf16<<<dim3(H / 32, NT / 32), dim3(32, 8), 0, stream>>>(
        (const float*)d_in[valIdx[i]], valT[i]);

  // Q / K / V pattentions fused into one launch (gridDim.y selects matrix)
  PattArgs pa;
  pa.key[0]  = keyb[0]; pa.key[1]  = keyb[1]; pa.key[2]  = keyb[2];
  pa.valT[0] = valT[0]; pa.valT[1] = valT[1]; pa.valT[2] = valT[2];
  pa.outB[0] = qb;      pa.outB[1] = kb;      pa.outB[2] = vb;
  pattention_kernel<false><<<dim3(S / 32, 3), 256, 0, stream>>>(xb, pa, nullptr);

  transpose_head<<<dim3(S / 64, NH), 256, 0, stream>>>(vb, vhT);

  flash_attn<<<dim3(S / 64, NH), 128, 0, stream>>>(qb, kb, vhT, ctx);

  // output projection pattention (fp32 output)
  PattArgs pp;
  pp.key[0]  = pp.key[1]  = pp.key[2]  = keyb[3];
  pp.valT[0] = pp.valT[1] = pp.valT[2] = valT[3];
  pp.outB[0] = pp.outB[1] = pp.outB[2] = nullptr;
  pattention_kernel<true><<<dim3(S / 32, 1), 256, 0, stream>>>(ctx, pp,
                                                               (float*)d_out);
}
